// PhysioGAT_64871186038939
// MI455X (gfx1250) — matrix operand order repaired and checked by reading, not compiled, not run on hardware
//
#include <hip/hip_runtime.h>
#include <hip/hip_bf16.h>
#include <math.h>

// ---------------------------------------------------------------------------
// PhysioGAT on MI455X (gfx1250):
//  - Layer-1 GAT is rank-1 (x is [N,1]) -> edge passes on scalars only.
//  - Layer-2 GEMM h1[N,512] x W2[512,128] via v_wmma_f32_16x16x32_f16.
//    Each wave owns TWO 16-row M-tiles; all 8 B fragments of a K-step are
//    preloaded into distinct registers, then 16 WMMAs issue back-to-back
//    (one dscnt wait per K-step, no WAR hazard NOPs inside the chain).
//  - Segment max via order-preserving uint encoding + atomicMax.
//  - Softmax denominators + layer-1 weighted sums fused in one edge pass.
// ---------------------------------------------------------------------------

typedef _Float16 half8 __attribute__((ext_vector_type(8)));
typedef _Float16 v16h  __attribute__((ext_vector_type(16)));
typedef float    v8f   __attribute__((ext_vector_type(8)));

#define NEG_SLOPE 0.2f
#define BN_EPS 1e-5f
#define ENC_NEG_INF 0x007FFFFFu   // encf(-inf)
#define GRAPHS 64

__device__ __forceinline__ unsigned encf(float f) {
  unsigned u = __float_as_uint(f);
  return (u & 0x80000000u) ? ~u : (u | 0x80000000u);
}
__device__ __forceinline__ float decf(unsigned u) {
  return (u & 0x80000000u) ? __uint_as_float(u & 0x7FFFFFFFu)
                           : __uint_as_float(~u);
}
__device__ __forceinline__ float lrelu(float v) {
  return v > 0.f ? v : NEG_SLOPE * v;
}

// cs1[h] = sum_f W1[h*64+f]*a1_src[h*64+f]; cd1 likewise with a1_dst.
__global__ void k_prep(const float* __restrict__ W1,
                       const float* __restrict__ a1s,
                       const float* __restrict__ a1d,
                       float* __restrict__ csd) {
  int t = threadIdx.x;
  if (t >= 16) return;
  int h = t & 7;
  const float* a = (t < 8) ? a1s : a1d;
  float acc = 0.f;
  for (int f = 0; f < 64; ++f) acc += W1[h * 64 + f] * a[h * 64 + f];
  csd[t] = acc;
}

__global__ void k_init(unsigned* __restrict__ m1, float* __restrict__ z1,
                       float* __restrict__ t1, unsigned* __restrict__ m2,
                       float* __restrict__ z2, float* __restrict__ out2,
                       float* __restrict__ pool, float* __restrict__ cnt,
                       int N) {
  int i = blockIdx.x * blockDim.x + threadIdx.x;
  if (i < N * 8)   { m1[i] = ENC_NEG_INF; z1[i] = 0.f; t1[i] = 0.f; }
  if (i < N * 4)   { m2[i] = ENC_NEG_INF; z2[i] = 0.f; }
  if (i < N * 128) { out2[i] = 0.f; }
  if (i < GRAPHS * 128) pool[i] = 0.f;
  if (i < GRAPHS)       cnt[i]  = 0.f;
}

__global__ void k_edge1_max(const float* __restrict__ x,
                            const int* __restrict__ ei, int E, int N,
                            const float* __restrict__ csd,
                            unsigned* __restrict__ m1) {
  int e = blockIdx.x * blockDim.x + threadIdx.x;
  if (e >= E + N) return;
  int s, d;
  if (e < E) { s = ei[e]; d = ei[E + e]; } else { s = d = e - E; }
  float xs = x[s], xd = x[d];
#pragma unroll
  for (int h = 0; h < 8; ++h) {
    float l = lrelu(xs * csd[h] + xd * csd[8 + h]);
    atomicMax(&m1[d * 8 + h], encf(l));
  }
}

// Fused: z1[d,h] += exp(l - m); t1[d,h] += exp(l - m) * x[src]
__global__ void k_edge1_sum(const float* __restrict__ x,
                            const int* __restrict__ ei, int E, int N,
                            const float* __restrict__ csd,
                            const unsigned* __restrict__ m1,
                            float* __restrict__ z1, float* __restrict__ t1) {
  int e = blockIdx.x * blockDim.x + threadIdx.x;
  if (e >= E + N) return;
  int s, d;
  if (e < E) { s = ei[e]; d = ei[E + e]; } else { s = d = e - E; }
  float xs = x[s], xd = x[d];
#pragma unroll
  for (int h = 0; h < 8; ++h) {
    float l = lrelu(xs * csd[h] + xd * csd[8 + h]);
    float ex = __expf(l - decf(m1[d * 8 + h]));
    atomicAdd(&z1[d * 8 + h], ex);
    atomicAdd(&t1[d * 8 + h], ex * xs);
  }
}

// h1[n,k] = relu( (t1/z1)[n,h]*W1[k] + b1[k] ), stored as f16.
__global__ void k_h1(const float* __restrict__ t1, const float* __restrict__ z1,
                     const float* __restrict__ W1, const float* __restrict__ b1,
                     _Float16* __restrict__ h1h, int N) {
  int i = blockIdx.x * blockDim.x + threadIdx.x;
  if (i >= N * 512) return;
  int n = i >> 9, k = i & 511, h = k >> 6;
  float s = t1[n * 8 + h] / z1[n * 8 + h];
  float v = fmaf(s, W1[k], b1[k]);
  h1h[i] = (_Float16)fmaxf(v, 0.f);
}

// wt[n*512+k] = (f16) W2[k*128+n]   (transpose + convert)
__global__ void k_wt(const float* __restrict__ W2, _Float16* __restrict__ wt) {
  int i = blockIdx.x * blockDim.x + threadIdx.x;
  if (i >= 128 * 512) return;
  int n = i >> 9, k = i & 511;
  wt[i] = (_Float16)W2[k * 128 + n];
}

// WMMA GEMM: C[N,128] = A[N,512](f16) x Wt^T.
// One wave per 32-row (two 16-row M-tiles) strip; 8 N-tiles; K stepped by 32.
// All 8 B fragments preloaded per K-step, then 16 back-to-back WMMAs.
__global__ void __launch_bounds__(128)
k_gemm(const _Float16* __restrict__ A, const _Float16* __restrict__ Wt,
       float* __restrict__ C, int Mtiles2) {
  extern __shared__ _Float16 sW[];  // 128*512 halves = 128 KB
  {
    const uint4* src = (const uint4*)Wt;
    uint4* dst = (uint4*)sW;
    for (int i = threadIdx.x; i < (128 * 512) / 8; i += blockDim.x)
      dst[i] = src[i];
  }
  __syncthreads();

  const int wave = threadIdx.x >> 5;
  const int lane = threadIdx.x & 31;
  const int mt2 = blockIdx.x * 4 + wave;
  if (mt2 >= Mtiles2) return;  // wave-uniform: EXEC stays all-ones for WMMA

  const int khalf = lane >> 4;        // A: 0 -> K 0..7/16..23 ; 1 -> 8..15/24..31
  const int ksel  = khalf * 8;        // B: 0 -> K 0..7/16..23 ; 1 -> 8..15/24..31
  const int col0  = lane & 15;

  const _Float16* arow0 = A + (size_t)(mt2 * 32 + (lane & 15)) * 512;
  const _Float16* arow1 = arow0 + (size_t)16 * 512;

  v8f acc0[8] = {};
  v8f acc1[8] = {};

  for (int kb = 0; kb < 512; kb += 32) {
    // ---- load both A fragments (global, b128 pairs) ----
    v16h a0, a1;
    {
      half8 lo = *(const half8*)(arow0 + kb + khalf * 8);
      half8 hi = *(const half8*)(arow0 + kb + 16 + khalf * 8);
#pragma unroll
      for (int i = 0; i < 8; ++i) { a0[i] = lo[i]; a0[i + 8] = hi[i]; }
      lo = *(const half8*)(arow1 + kb + khalf * 8);
      hi = *(const half8*)(arow1 + kb + 16 + khalf * 8);
#pragma unroll
      for (int i = 0; i < 8; ++i) { a1[i] = lo[i]; a1[i + 8] = hi[i]; }
    }

    // ---- preload all 8 B fragments into distinct registers ----
    v16h bf[8];
#pragma unroll
    for (int nt = 0; nt < 8; ++nt) {
      const _Float16* bcol = sW + (size_t)(nt * 16 + col0) * 512 + kb + ksel;
      half8 blo = *(const half8*)(bcol);
      half8 bhi = *(const half8*)(bcol + 16);
#pragma unroll
      for (int i = 0; i < 8; ++i) { bf[nt][i] = blo[i]; bf[nt][i + 8] = bhi[i]; }
    }

    // ---- 16 back-to-back WMMAs (no loads interleaved) ----
#pragma unroll
    for (int nt = 0; nt < 8; ++nt) {
      acc0[nt] = __builtin_amdgcn_wmma_f32_16x16x32_f16(
          false, a0, false, bf[nt], (short)0, acc0[nt], false, false);
      acc1[nt] = __builtin_amdgcn_wmma_f32_16x16x32_f16(
          false, a1, false, bf[nt], (short)0, acc1[nt], false, false);
    }
  }

  const int rb0 = mt2 * 32 + khalf * 8;  // C layout rows, tile 0
  const int rb1 = rb0 + 16;              // tile 1
  const int cb = lane & 15;
#pragma unroll
  for (int nt = 0; nt < 8; ++nt) {
#pragma unroll
    for (int v = 0; v < 8; ++v) {
      C[(size_t)(rb0 + v) * 128 + nt * 16 + cb] = acc0[nt][v];
      C[(size_t)(rb1 + v) * 128 + nt * 16 + cb] = acc1[nt][v];
    }
  }
}

// alpha2 src/dst logits: dot(h2pre[n,h,0:32], a2_{src,dst}[h,0:32])
__global__ void k_alpha2(const float* __restrict__ h2,
                         const float* __restrict__ a2s,
                         const float* __restrict__ a2d,
                         float* __restrict__ as2, float* __restrict__ ad2,
                         int N) {
  int i = blockIdx.x * blockDim.x + threadIdx.x;
  if (i >= N * 4) return;
  int n = i >> 2, h = i & 3;
  const float* hp = h2 + (size_t)n * 128 + h * 32;
  const float* as = a2s + h * 32;
  const float* ad = a2d + h * 32;
  float s = 0.f, d = 0.f;
#pragma unroll
  for (int f = 0; f < 32; ++f) {
    float v = hp[f];
    s = fmaf(v, as[f], s);
    d = fmaf(v, ad[f], d);
  }
  as2[i] = s;
  ad2[i] = d;
}

__global__ void k_edge2_max(const int* __restrict__ ei, int E, int N,
                            const float* __restrict__ as2,
                            const float* __restrict__ ad2,
                            unsigned* __restrict__ m2) {
  int e = blockIdx.x * blockDim.x + threadIdx.x;
  if (e >= E + N) return;
  int s, d;
  if (e < E) { s = ei[e]; d = ei[E + e]; } else { s = d = e - E; }
#pragma unroll
  for (int h = 0; h < 4; ++h) {
    float l = lrelu(as2[s * 4 + h] + ad2[d * 4 + h]);
    atomicMax(&m2[d * 4 + h], encf(l));
  }
}

__global__ void k_edge2_z(const int* __restrict__ ei, int E, int N,
                          const float* __restrict__ as2,
                          const float* __restrict__ ad2,
                          const unsigned* __restrict__ m2,
                          float* __restrict__ z2) {
  int e = blockIdx.x * blockDim.x + threadIdx.x;
  if (e >= E + N) return;
  int s, d;
  if (e < E) { s = ei[e]; d = ei[E + e]; } else { s = d = e - E; }
#pragma unroll
  for (int h = 0; h < 4; ++h) {
    float l = lrelu(as2[s * 4 + h] + ad2[d * 4 + h]);
    atomicAdd(&z2[d * 4 + h], __expf(l - decf(m2[d * 4 + h])));
  }
}

// out2[dst, h*32+f] += alpha * h2pre[src, h*32+f]; one thread per (edge, head)
__global__ void k_edge2_agg(const int* __restrict__ ei, int E, int N,
                            const float* __restrict__ as2,
                            const float* __restrict__ ad2,
                            const unsigned* __restrict__ m2,
                            const float* __restrict__ z2,
                            const float* __restrict__ h2,
                            float* __restrict__ out2) {
  int i = blockIdx.x * blockDim.x + threadIdx.x;
  if (i >= (E + N) * 4) return;
  int e = i >> 2, h = i & 3;
  int s, d;
  if (e < E) { s = ei[e]; d = ei[E + e]; } else { s = d = e - E; }
  float l = lrelu(as2[s * 4 + h] + ad2[d * 4 + h]);
  float alpha = __expf(l - decf(m2[d * 4 + h])) / z2[d * 4 + h];
  const float4* hp = (const float4*)(h2 + (size_t)s * 128 + h * 32);
  float* op = out2 + (size_t)d * 128 + h * 32;
#pragma unroll
  for (int q = 0; q < 8; ++q) {
    float4 v = hp[q];
    atomicAdd(&op[q * 4 + 0], alpha * v.x);
    atomicAdd(&op[q * 4 + 1], alpha * v.y);
    atomicAdd(&op[q * 4 + 2], alpha * v.z);
    atomicAdd(&op[q * 4 + 3], alpha * v.w);
  }
}

// relu(out2 + b2), accumulate per-graph sums and counts.
__global__ void k_pool(const float* __restrict__ out2,
                       const float* __restrict__ b2,
                       const int* __restrict__ batch,
                       float* __restrict__ pool, float* __restrict__ cnt,
                       int N) {
  int i = blockIdx.x * blockDim.x + threadIdx.x;
  if (i >= N * 8) return;
  int n = i >> 3, c = i & 7;
  int g = batch[n];
  int f0 = c * 16;
#pragma unroll
  for (int f = f0; f < f0 + 16; ++f) {
    float v = out2[(size_t)n * 128 + f] + b2[f];
    atomicAdd(&pool[g * 128 + f], fmaxf(v, 0.f));
  }
  if (c == 0) atomicAdd(&cnt[g], 1.0f);
}

// Single block: pooled -> FC -> BN(batch stats over 64) -> relu -> heads.
__global__ void __launch_bounds__(256)
k_head(const float* __restrict__ pool, const float* __restrict__ cnt,
       const float* __restrict__ fcW, const float* __restrict__ fcb,
       const float* __restrict__ gamma, const float* __restrict__ beta,
       const float* __restrict__ hW, const float* __restrict__ hb,
       const float* __restrict__ dW, const float* __restrict__ db,
       float* __restrict__ out) {
  __shared__ float pooled[GRAPHS * 128];
  __shared__ float yb[GRAPHS * 16];
  __shared__ float mu[16], iv[16];
  int t = threadIdx.x;

  for (int i = t; i < GRAPHS * 128; i += 256) {
    int g = i >> 7;
    pooled[i] = pool[i] / fmaxf(cnt[g], 1.0f);
  }
  __syncthreads();

  for (int i = t; i < GRAPHS * 16; i += 256) {
    int g = i >> 4, j = i & 15;
    float acc = fcb[j];
    for (int f = 0; f < 128; ++f)
      acc = fmaf(pooled[g * 128 + f], fcW[f * 16 + j], acc);
    yb[i] = acc;
  }
  __syncthreads();

  if (t < 16) {
    float m = 0.f;
    for (int g = 0; g < GRAPHS; ++g) m += yb[g * 16 + t];
    m /= (float)GRAPHS;
    float v = 0.f;
    for (int g = 0; g < GRAPHS; ++g) {
      float dlt = yb[g * 16 + t] - m;
      v += dlt * dlt;
    }
    v /= (float)GRAPHS;
    mu[t] = m;
    iv[t] = rsqrtf(v + BN_EPS);
  }
  __syncthreads();

  for (int i = t; i < GRAPHS * 16; i += 256) {
    int j = i & 15;
    float v = (yb[i] - mu[j]) * iv[j] * gamma[j] + beta[j];
    yb[i] = fmaxf(v, 0.f);
  }
  __syncthreads();

  for (int i = t; i < GRAPHS * 3; i += 256) {
    int g = i / 3, j = i % 3;
    float acc = hb[j];
    for (int k = 0; k < 16; ++k) acc = fmaf(yb[g * 16 + k], hW[k * 3 + j], acc);
    out[i] = acc;
  }
  for (int i = t; i < GRAPHS * 2; i += 256) {
    int g = i >> 1, j = i & 1;
    float acc = db[j];
    for (int k = 0; k < 16; ++k) acc = fmaf(yb[g * 16 + k], dW[k * 2 + j], acc);
    out[GRAPHS * 3 + i] = acc;
  }
}

extern "C" void kernel_launch(void* const* d_in, const int* in_sizes, int n_in,
                              void* d_out, int out_size, void* d_ws,
                              size_t ws_size, hipStream_t stream) {
  (void)n_in; (void)out_size; (void)ws_size;
  const float* x     = (const float*)d_in[0];
  const int*   ei    = (const int*)d_in[1];
  const int*   batch = (const int*)d_in[2];
  const float* W1    = (const float*)d_in[3];
  const float* a1s   = (const float*)d_in[4];
  const float* a1d   = (const float*)d_in[5];
  const float* b1    = (const float*)d_in[6];
  const float* W2    = (const float*)d_in[7];
  const float* a2s   = (const float*)d_in[8];
  const float* a2d   = (const float*)d_in[9];
  const float* b2    = (const float*)d_in[10];
  const float* fcW   = (const float*)d_in[11];
  const float* fcb   = (const float*)d_in[12];
  const float* gamma = (const float*)d_in[13];
  const float* beta  = (const float*)d_in[14];
  const float* hW    = (const float*)d_in[15];
  const float* hb    = (const float*)d_in[16];
  const float* dW    = (const float*)d_in[17];
  const float* db    = (const float*)d_in[18];
  float* out = (float*)d_out;

  const int N  = in_sizes[0];       // 20000 (multiple of 32)
  const int E  = in_sizes[1] / 2;   // 160000
  const int EN = E + N;             // edges + self-loops

  // Workspace carve-up (≈45 MB total)
  char* base = (char*)d_ws;
  size_t off = 0;
  auto take = [&](size_t bytes) -> char* {
    char* p = base + off;
    off += (bytes + 255) & ~(size_t)255;
    return p;
  };
  float*    csd  = (float*)take(16 * 4);
  unsigned* m1   = (unsigned*)take((size_t)N * 8 * 4);
  float*    z1   = (float*)take((size_t)N * 8 * 4);
  float*    t1   = (float*)take((size_t)N * 8 * 4);
  float*    as2v = (float*)take((size_t)N * 4 * 4);
  float*    ad2v = (float*)take((size_t)N * 4 * 4);
  unsigned* m2   = (unsigned*)take((size_t)N * 4 * 4);
  float*    z2   = (float*)take((size_t)N * 4 * 4);
  _Float16* h1h  = (_Float16*)take((size_t)N * 512 * 2);
  _Float16* wt   = (_Float16*)take((size_t)128 * 512 * 2);
  float*    h2p  = (float*)take((size_t)N * 128 * 4);
  float*    out2 = (float*)take((size_t)N * 128 * 4);
  float*    pool = (float*)take((size_t)GRAPHS * 128 * 4);
  float*    cnt  = (float*)take((size_t)GRAPHS * 4);

  dim3 b256(256);
  k_prep<<<1, 32, 0, stream>>>(W1, a1s, a1d, csd);
  k_init<<<(N * 128 + 255) / 256, b256, 0, stream>>>(m1, z1, t1, m2, z2, out2,
                                                     pool, cnt, N);
  k_edge1_max<<<(EN + 255) / 256, b256, 0, stream>>>(x, ei, E, N, csd, m1);
  k_edge1_sum<<<(EN + 255) / 256, b256, 0, stream>>>(x, ei, E, N, csd, m1, z1,
                                                     t1);
  k_h1<<<(N * 512 + 255) / 256, b256, 0, stream>>>(t1, z1, W1, b1, h1h, N);
  k_wt<<<(128 * 512 + 255) / 256, b256, 0, stream>>>(W2, wt);

  const int Mtiles2 = N / 32;  // 32 rows per wave (two 16-row WMMA tiles)
  k_gemm<<<(Mtiles2 + 3) / 4, dim3(128), 128 * 512 * 2, stream>>>(h1h, wt, h2p,
                                                                  Mtiles2);

  k_alpha2<<<(N * 4 + 255) / 256, b256, 0, stream>>>(h2p, a2s, a2d, as2v, ad2v,
                                                     N);
  k_edge2_max<<<(EN + 255) / 256, b256, 0, stream>>>(ei, E, N, as2v, ad2v, m2);
  k_edge2_z<<<(EN + 255) / 256, b256, 0, stream>>>(ei, E, N, as2v, ad2v, m2,
                                                   z2);
  k_edge2_agg<<<(EN * 4 + 255) / 256, b256, 0, stream>>>(ei, E, N, as2v, ad2v,
                                                         m2, z2, h2p, out2);
  k_pool<<<(N * 8 + 255) / 256, b256, 0, stream>>>(out2, b2, batch, pool, cnt,
                                                   N);
  k_head<<<1, 256, 0, stream>>>(pool, cnt, fcW, fcb, gamma, beta, hW, hb, dW,
                                db, out);
}
